// HRAMPolicyNetDoc_87488483819538
// MI455X (gfx1250) — matrix operand order repaired and checked by reading, not compiled, not run on hardware
//
#include <hip/hip_runtime.h>
#include <hip/hip_bf16.h>
#include <math.h>

// ---------------------------------------------------------------------------
// MI455X (gfx1250): all GEMMs on v_wmma_f32_16x16x32_f16 (f16 in, f32 acc).
// Dominant cost: qn @ kn^T (8192x3072x4096, ~206 GFLOP) -> compute bound.
// Scores GEMM: double-buffered async global->LDS (ASYNCcnt) pipeline, no
// guards (exact tile multiples). Other GEMMs: generic guarded f32 kernel.
// ---------------------------------------------------------------------------

typedef __attribute__((ext_vector_type(16))) _Float16 v16h;
typedef __attribute__((ext_vector_type(8)))  float    v8f;
typedef __attribute__((ext_vector_type(8)))  _Float16 h8;
typedef __attribute__((ext_vector_type(4)))  _Float16 h4;
typedef __attribute__((ext_vector_type(4)))  float    f4;
typedef __attribute__((ext_vector_type(4)))  int      v4i;

union Frag16 { v16h v; _Float16 h[16]; };
union Frag8f { v8f  v; float     f[8];  };

#define TM 128
#define TN 64
#define TK 32

#define AS1 __attribute__((address_space(1)))
#define AS3 __attribute__((address_space(3)))

#if __has_builtin(__builtin_amdgcn_global_load_async_to_lds_b128)
#define HAVE_ASYNC_LDS 1
#endif

#if __has_builtin(__builtin_amdgcn_s_wait_asynccnt)
#define ASYNC_WAIT() __builtin_amdgcn_s_wait_asynccnt(0)
#else
#define ASYNC_WAIT() asm volatile("s_wait_asynccnt 0x0" ::: "memory")
#endif

// ---------------------------------------------------------------------------
// Scores GEMM: Y(MxN) = Xh(MxK) @ Wh(NxK)^T, f16 inputs, f32 out.
// Requires M%TM==0, N%TN==0, K%TK==0 (true for 8192x4096x3072).
// Double-buffered LDS; tile k+1 streams in via async global->LDS while the
// WMMAs consume tile k; one barrier per K-step.
// ---------------------------------------------------------------------------
__launch_bounds__(256)
__global__ void gemm_f16t_k(const _Float16* __restrict__ Xh,
                            const _Float16* __restrict__ Wh,
                            float* __restrict__ Y, int M, int N, int Kd)
{
    __shared__ _Float16 xs[2][TM][TK];   // 2 x 8 KB
    __shared__ _Float16 ws[2][TN][TK];   // 2 x 4 KB

    const int tid  = threadIdx.x;
    const int wave = tid >> 5;
    const int lane = tid & 31;
    const int bm = blockIdx.y * TM;
    const int bn = blockIdx.x * TN;

    auto stage = [&](int buf, int k0) {
#if defined(HAVE_ASYNC_LDS)
        #pragma unroll
        for (int j = 0; j < 2; ++j) {
            int c = tid + (j << 8);            // 0..511
            int r = c >> 2, k8 = (c & 3) << 3;
            __builtin_amdgcn_global_load_async_to_lds_b128(
                (AS1 v4i*)(Xh + (size_t)(bm + r) * Kd + k0 + k8),
                (AS3 v4i*)(&xs[buf][r][k8]), 0, 0);
        }
        {
            int n = tid >> 2, k8 = (tid & 3) << 3;
            __builtin_amdgcn_global_load_async_to_lds_b128(
                (AS1 v4i*)(Wh + (size_t)(bn + n) * Kd + k0 + k8),
                (AS3 v4i*)(&ws[buf][n][k8]), 0, 0);
        }
#else
        #pragma unroll
        for (int j = 0; j < 2; ++j) {
            int c = tid + (j << 8);
            int r = c >> 2, k8 = (c & 3) << 3;
            *(h8*)&xs[buf][r][k8] = *(const h8*)(Xh + (size_t)(bm + r) * Kd + k0 + k8);
        }
        {
            int n = tid >> 2, k8 = (tid & 3) << 3;
            *(h8*)&ws[buf][n][k8] = *(const h8*)(Wh + (size_t)(bn + n) * Kd + k0 + k8);
        }
#endif
    };

    Frag8f acc[4];
    #pragma unroll
    for (int t = 0; t < 4; ++t)
        #pragma unroll
        for (int j = 0; j < 8; ++j) acc[t].f[j] = 0.f;

    stage(0, 0);
    ASYNC_WAIT();
    __syncthreads();

    const int nk = Kd / TK;
    const int mrow  = lane & 15;
    const int khalf = lane >> 4;

    for (int it = 0; it < nk; ++it) {
        int buf = it & 1;
        if (it + 1 < nk) stage(buf ^ 1, (it + 1) * TK);   // overlap with compute

        Frag16 af;
        #pragma unroll
        for (int j = 0; j < 8; ++j) {
            int kk = ((j & 4) ? 16 : 0) + (j & 3) * 2 + khalf * 8;
            af.h[2 * j]     = xs[buf][wave * 16 + mrow][kk];
            af.h[2 * j + 1] = xs[buf][wave * 16 + mrow][kk + 1];
        }
        #pragma unroll
        for (int t = 0; t < 4; ++t) {
            Frag16 bf;
            int ncol = t * 16 + (lane & 15);
            #pragma unroll
            for (int j = 0; j < 8; ++j) {
                int kk = ((j & 4) ? 16 : 0) + (j & 3) * 2 + khalf * 8;
                bf.h[2 * j]     = ws[buf][ncol][kk];
                bf.h[2 * j + 1] = ws[buf][ncol][kk + 1];
            }
            acc[t].v = __builtin_amdgcn_wmma_f32_16x16x32_f16(
                false, af.v, false, bf.v, (short)0, acc[t].v, false, false);
        }
        ASYNC_WAIT();          // next tile resident before anyone reads it
        __syncthreads();
    }

    const int ncl   = lane & 15;
    const int mhalf = lane >> 4;
    #pragma unroll
    for (int t = 0; t < 4; ++t) {
        int gn = bn + t * 16 + ncl;
        #pragma unroll
        for (int j = 0; j < 8; ++j) {
            int gm = bm + wave * 16 + mhalf * 8 + j;
            Y[(size_t)gm * N + gn] = acc[t].f[j];
        }
    }
}

// ---------------------------------------------------------------------------
// Generic f32 GEMM: Y = act(X @ W + bias), W is (K,N) row-major.
// 256 threads = 8 waves; block tile 128x64; wave tile 16x64 (4 accumulators).
// act: 0=none, 1=relu, 2=clip(+-20).
// ---------------------------------------------------------------------------
__launch_bounds__(256)
__global__ void gemm_k(const float* __restrict__ Xf, const float* __restrict__ Wf,
                       const float* __restrict__ bias, float* __restrict__ Y,
                       int M, int N, int Kd, int act)
{
    __shared__ _Float16 xs[TM][TK];   // 8 KB
    __shared__ _Float16 ws[TN][TK];   // 4 KB, [n][k] so B-frag reads are contiguous

    const int tid  = threadIdx.x;
    const int wave = tid >> 5;
    const int lane = tid & 31;
    const int bm = blockIdx.y * TM;
    const int bn = blockIdx.x * TN;

    Frag8f acc[4];
    #pragma unroll
    for (int t = 0; t < 4; ++t)
        #pragma unroll
        for (int j = 0; j < 8; ++j) acc[t].f[j] = 0.f;

    const bool fullMN = (bm + TM <= M) && (bn + TN <= N);

    for (int k0 = 0; k0 < Kd; k0 += TK) {
        const bool fullTile = fullMN && (k0 + TK <= Kd);   // wave-uniform
        if (fullTile) {
            // b128 loads, convert, wide LDS stores
            #pragma unroll
            for (int j = 0; j < 4; ++j) {
                int c = tid + (j << 8);            // 0..1023
                int r = c >> 3, c4 = (c & 7) << 2;
                f4 v = *(const f4*)(Xf + (size_t)(bm + r) * Kd + k0 + c4);
                *(h4*)&xs[r][c4] = __builtin_convertvector(v, h4);
            }
            #pragma unroll
            for (int j = 0; j < 2; ++j) {
                int c = tid + (j << 8);            // 0..511
                int k = c >> 4, n4 = (c & 15) << 2;
                f4 v = *(const f4*)(Wf + (size_t)(k0 + k) * N + bn + n4);
                ws[n4 + 0][k] = (_Float16)v.x;
                ws[n4 + 1][k] = (_Float16)v.y;
                ws[n4 + 2][k] = (_Float16)v.z;
                ws[n4 + 3][k] = (_Float16)v.w;
            }
            if (k0 + TK < Kd)
                __builtin_prefetch(&Xf[(size_t)(bm + (tid >> 1)) * Kd + k0 + TK], 0, 1);
        } else {
            // guarded scalar staging (remainder K tiles / small-N blocks only)
            for (int i = tid; i < TM * TK; i += 256) {
                int r = i >> 5, c = i & 31;
                int gm = bm + r, gk = k0 + c;
                float fv = (gk < Kd && gm < M) ? Xf[(size_t)gm * Kd + gk] : 0.f;
                xs[r][c] = (_Float16)fv;
            }
            for (int i = tid; i < TN * TK; i += 256) {
                int n = i >> 5, c = i & 31;
                int gn = bn + n, gk = k0 + c;
                float fv = (gk < Kd && gn < N) ? Wf[(size_t)gk * N + gn] : 0.f;
                ws[n][c] = (_Float16)fv;
            }
        }
        __syncthreads();

        // ---- A fragment (16x32 f16, ISA layout) ----
        const int mrow  = lane & 15;
        const int khalf = lane >> 4;
        Frag16 af;
        #pragma unroll
        for (int j = 0; j < 8; ++j) {
            int kk = ((j & 4) ? 16 : 0) + (j & 3) * 2 + khalf * 8;
            af.h[2 * j]     = xs[wave * 16 + mrow][kk];
            af.h[2 * j + 1] = xs[wave * 16 + mrow][kk + 1];
        }
        // ---- 4 x (B frag + WMMA) ----
        #pragma unroll
        for (int t = 0; t < 4; ++t) {
            Frag16 bf;
            int ncol = t * 16 + (lane & 15);
            #pragma unroll
            for (int j = 0; j < 8; ++j) {
                int kk = ((j & 4) ? 16 : 0) + (j & 3) * 2 + khalf * 8;
                bf.h[2 * j]     = ws[ncol][kk];
                bf.h[2 * j + 1] = ws[ncol][kk + 1];
            }
            acc[t].v = __builtin_amdgcn_wmma_f32_16x16x32_f16(
                false, af.v, false, bf.v, (short)0, acc[t].v, false, false);
        }
        __syncthreads();
    }

    // ---- epilogue: C layout m = vgpr + 8*(lane/16), n = lane%16 ----
    const int ncl   = lane & 15;
    const int mhalf = lane >> 4;
    #pragma unroll
    for (int t = 0; t < 4; ++t) {
        int gn = bn + t * 16 + ncl;
        if (gn < N) {
            float bv = bias ? bias[gn] : 0.f;
            #pragma unroll
            for (int j = 0; j < 8; ++j) {
                int gm = bm + wave * 16 + mhalf * 8 + j;
                if (gm < M) {
                    float v = acc[t].f[j] + bv;
                    if (act == 1)      v = fmaxf(v, 0.f);
                    else if (act == 2) v = fminf(fmaxf(v, -20.f), 20.f);
                    Y[(size_t)gm * N + gn] = v;
                }
            }
        }
    }
}

// ---------------------------------------------------------------------------
// Row-wise kernels
// ---------------------------------------------------------------------------
__device__ inline float bsum256(float v, float* red)
{
    int tid = threadIdx.x;
    red[tid] = v; __syncthreads();
    for (int s = 128; s > 0; s >>= 1) {
        if (tid < s) red[tid] += red[tid + s];
        __syncthreads();
    }
    float r = red[0];
    __syncthreads();
    return r;
}

// Optional LN then L2-normalize a row (len multiple of 256, <= 3072) -> f16.
__launch_bounds__(256)
__global__ void rownorm_f16_k(const float* __restrict__ X, const float* __restrict__ g,
                              const float* __restrict__ be, _Float16* __restrict__ Yh,
                              int rowlen, int do_ln)
{
    __shared__ float red[256];
    int row = blockIdx.x, tid = threadIdx.x;
    int nv = rowlen >> 8;
    float v[12];
    const float* xr = X + (size_t)row * rowlen;
    for (int i = 0; i < nv; ++i) v[i] = xr[tid + (i << 8)];
    if (do_ln) {
        float s = 0.f;
        for (int i = 0; i < nv; ++i) s += v[i];
        float mean = bsum256(s, red) / (float)rowlen;
        float q = 0.f;
        for (int i = 0; i < nv; ++i) { float d = v[i] - mean; q += d * d; }
        float var = bsum256(q, red) / (float)rowlen;
        float rs = rsqrtf(var + 1e-5f);
        for (int i = 0; i < nv; ++i) {
            int c = tid + (i << 8);
            v[i] = (v[i] - mean) * rs * g[c] + be[c];
        }
    }
    float ss = 0.f;
    for (int i = 0; i < nv; ++i) ss += v[i] * v[i];
    float tot = bsum256(ss, red);
    float inv = rsqrtf(tot + 1e-12f);
    _Float16* yr = Yh + (size_t)row * rowlen;
    for (int i = 0; i < nv; ++i) yr[tid + (i << 8)] = (_Float16)(v[i] * inv);
}

// In-place LayerNorm over 128-wide rows, optional relu.
__launch_bounds__(128)
__global__ void ln128_k(float* __restrict__ X, const float* __restrict__ g,
                        const float* __restrict__ be, int relu)
{
    __shared__ float red[128];
    int row = blockIdx.x, tid = threadIdx.x;
    float v = X[(size_t)row * 128 + tid];
    red[tid] = v; __syncthreads();
    for (int s = 64; s > 0; s >>= 1) { if (tid < s) red[tid] += red[tid + s]; __syncthreads(); }
    float mean = red[0] / 128.f; __syncthreads();
    float d = v - mean;
    red[tid] = d * d; __syncthreads();
    for (int s = 64; s > 0; s >>= 1) { if (tid < s) red[tid] += red[tid + s]; __syncthreads(); }
    float var = red[0] / 128.f; __syncthreads();
    float y = d * rsqrtf(var + 1e-5f) * g[tid] + be[tid];
    if (relu) y = fmaxf(y, 0.f);
    X[(size_t)row * 128 + tid] = y;
}

// top-10 over 4096 scores per row; scores kept in registers (16/lane).
__launch_bounds__(256)
__global__ void topk10_k(const float* __restrict__ SC, int* __restrict__ IDX)
{
    __shared__ float sv[256];
    __shared__ int   si[256];
    int row = blockIdx.x, tid = threadIdx.x;
    const float* sr = SC + (size_t)row * 4096;
    float lv[16];
    #pragma unroll
    for (int s = 0; s < 16; ++s) lv[s] = sr[tid + (s << 8)];
    for (int sel = 0; sel < 10; ++sel) {
        float bv = -__builtin_inff(); int bi = 0x7fffffff;
        #pragma unroll
        for (int s = 0; s < 16; ++s) {
            int col = tid + (s << 8);
            if (lv[s] > bv) { bv = lv[s]; bi = col; }
        }
        sv[tid] = bv; si[tid] = bi; __syncthreads();
        for (int s = 128; s > 0; s >>= 1) {
            if (tid < s) {
                float ov = sv[tid + s]; int oi = si[tid + s];
                if (ov > sv[tid] || (ov == sv[tid] && oi < si[tid])) { sv[tid] = ov; si[tid] = oi; }
            }
            __syncthreads();
        }
        int best = si[0]; __syncthreads();
        if ((best & 255) == tid) lv[best >> 8] = -__builtin_inff();
        if (tid == 0) IDX[row * 10 + sel] = best;
    }
}

// context = mean of 10 selected key rows (raw f32 keys)
__launch_bounds__(256)
__global__ void context_k(const float* __restrict__ keys, const int* __restrict__ IDX,
                          float* __restrict__ ctx)
{
    __shared__ int sidx[10];
    int row = blockIdx.x, tid = threadIdx.x;
    if (tid < 10) sidx[tid] = IDX[row * 10 + tid];
    __syncthreads();
    for (int d = tid; d < 3072; d += 256) {
        float s = 0.f;
        #pragma unroll
        for (int j = 0; j < 10; ++j) s += keys[(size_t)sidx[j] * 3072 + d];
        ctx[(size_t)row * 3072 + d] = s / 10.0f;
    }
}

// XC = [shared(128) | state[:, off:off+len]]
__global__ void concat_k(const float* __restrict__ SH, const float* __restrict__ state,
                         float* __restrict__ XC, int wid, int off, int total)
{
    int i = blockIdx.x * blockDim.x + threadIdx.x;
    if (i >= total) return;
    int row = i / wid, c = i % wid;
    float v = (c < 128) ? SH[(size_t)row * 128 + c]
                        : state[(size_t)row * 115 + off + (c - 128)];
    XC[(size_t)row * wid + c] = v;
}

// gumbel softmax (tau=1), straight-through alpha, and logits mix
__launch_bounds__(256)
__global__ void alpha_logits_k(const float* __restrict__ RL, const float* __restrict__ gum,
                               const float* __restrict__ LP, float* __restrict__ out_logits,
                               float* __restrict__ out_alpha)
{
    int row = blockIdx.x * blockDim.x + threadIdx.x;
    if (row >= 8192) return;
    float y[4], soft[4];
    float mx = -__builtin_inff();
    #pragma unroll
    for (int k = 0; k < 4; ++k) {
        y[k] = RL[row * 4 + k] + gum[row * 4 + k];   // TAU = 1.0
        mx = fmaxf(mx, y[k]);
    }
    float s = 0.f;
    #pragma unroll
    for (int k = 0; k < 4; ++k) { soft[k] = expf(y[k] - mx); s += soft[k]; }
    #pragma unroll
    for (int k = 0; k < 4; ++k) soft[k] /= s;
    int am = 0; float bv = soft[0];
    #pragma unroll
    for (int k = 1; k < 4; ++k) if (soft[k] > bv) { bv = soft[k]; am = k; }
    float alpha[4];
    #pragma unroll
    for (int k = 0; k < 4; ++k) {
        float hard = (k == am) ? 1.f : 0.f;
        alpha[k] = soft[k] + (hard - soft[k]);   // straight-through, literal
        out_alpha[row * 4 + k] = alpha[k];
    }
    for (int a = 0; a < 23; ++a) {
        float acc = 0.f;
        #pragma unroll
        for (int k = 0; k < 4; ++k)
            acc += alpha[k] * LP[((size_t)k * 8192 + row) * 23 + a];
        out_logits[row * 23 + a] = acc;
    }
}

// ---------------------------------------------------------------------------
// Host launch sequence
// ---------------------------------------------------------------------------
extern "C" void kernel_launch(void* const* d_in, const int* in_sizes, int n_in,
                              void* d_out, int out_size, void* d_ws, size_t ws_size,
                              hipStream_t stream)
{
    (void)in_sizes; (void)n_in; (void)out_size; (void)ws_size;
#define IN(i) ((const float*)d_in[i])
    const float* state  = IN(0);
    const float* gumbel = IN(1);
    const float* keys   = IN(2);

    // workspace layout (REG0 time-shared: q -> scores -> context)
    char* wsb = (char*)d_ws;
    float*    REG0 = (float*)(wsb + 0ull);             // 128 MB
    _Float16* QH   = (_Float16*)(wsb + 134217728ull);  // 48 MB  (8192x3072 f16)
    _Float16* KNH  = (_Float16*)(wsb + 184549376ull);  // 24 MB  (4096x3072 f16)
    float*    T1   = (float*)(wsb + 209715200ull);     // 16 MB
    float*    T2   = (float*)(wsb + 226492416ull);     // 16 MB (A1/A2/C2 reuse)
    float*    SH   = (float*)(wsb + 243269632ull);     // 4 MB  (8192x128)
    float*    XC   = (float*)(wsb + 247463936ull);     // 6 MB  (8192x192 max)
    float*    LP   = (float*)(wsb + 253755392ull);     // 4x8192x23
    float*    R1   = (float*)(wsb + 256770048ull);     // 8192x64
    float*    R2   = (float*)(wsb + 258867200ull);     // 8192x64
    float*    RL   = (float*)(wsb + 260964352ull);     // 8192x4
    int*      IDX  = (int*)  (wsb + 261095424ull);     // 8192x10
    float* A1 = T2;
    float* A2 = T2 + (size_t)8192 * 128;

    float* out        = (float*)d_out;
    float* out_logits = out;                       // 8192x23
    float* out_alpha  = out + (size_t)8192 * 23;   // 8192x4
    float* out_value  = out + (size_t)8192 * 27;   // 8192x1

    auto gemm = [&](const float* X, const float* W, const float* b, float* Y,
                    int M, int N, int K, int act) {
        dim3 g((N + TN - 1) / TN, (M + TM - 1) / TM);
        gemm_k<<<g, 256, 0, stream>>>(X, W, b, Y, M, N, K, act);
    };

    // ---- state encoder: 115 -> 512 -> 512 -> 3072, then LN + L2-norm -> f16 ----
    gemm(state, IN(3), IN(4), T1,   8192, 512,  115, 1);
    gemm(T1,    IN(5), IN(6), T2,   8192, 512,  512, 1);
    gemm(T2,    IN(7), IN(8), REG0, 8192, 3072, 512, 0);            // q (pre-LN)
    rownorm_f16_k<<<8192, 256, 0, stream>>>(REG0, IN(9), IN(10), QH, 3072, 1);     // qn
    rownorm_f16_k<<<4096, 256, 0, stream>>>(keys, nullptr, nullptr, KNH, 3072, 0); // kn

    // ---- scores = qn @ kn^T (dominant GEMM, async double-buffered WMMA) ----
    {
        dim3 gs(4096 / TN, 8192 / TM);
        gemm_f16t_k<<<gs, 256, 0, stream>>>(QH, KNH, REG0, 8192, 4096, 3072);
    }

    // ---- top-10 + context mean ----
    topk10_k<<<8192, 256, 0, stream>>>(REG0, IDX);
    context_k<<<8192, 256, 0, stream>>>(keys, IDX, REG0);           // context reuses REG0

    // ---- context compressor: 3072 -> 512 -> 128, LN ----
    gemm(REG0, IN(11), IN(12), T1, 8192, 512, 3072, 1);
    gemm(T1,   IN(13), IN(14), SH, 8192, 128, 512,  0);
    ln128_k<<<8192, 128, 0, stream>>>(SH, IN(15), IN(16), 0);       // shared

    // ---- four actor heads ----
    const int abase[4] = {17, 27, 37, 47};
    const int aoff[4]  = {0, 20, 40, 45};
    const int alen[4]  = {50, 20, 10, 15};
    for (int p = 0; p < 4; ++p) {
        int b0 = abase[p];
        int wid = 128 + alen[p];
        int total = 8192 * wid;
        concat_k<<<(total + 255) / 256, 256, 0, stream>>>(SH, state, XC, wid, aoff[p], total);
        gemm(XC, IN(b0 + 0), IN(b0 + 1), A1, 8192, 128, wid, 0);
        ln128_k<<<8192, 128, 0, stream>>>(A1, IN(b0 + 2), IN(b0 + 3), 1);
        gemm(A1, IN(b0 + 4), IN(b0 + 5), A2, 8192, 128, 128, 0);
        ln128_k<<<8192, 128, 0, stream>>>(A2, IN(b0 + 6), IN(b0 + 7), 1);
        gemm(A2, IN(b0 + 8), IN(b0 + 9), LP + (size_t)p * 8192 * 23, 8192, 23, 128, 0);
    }

    // ---- router: 115 -> 64 -> 64 -> 4 (clip +-20), then gumbel/ST + mix ----
    gemm(state, IN(57), IN(58), R1, 8192, 64, 115, 1);
    gemm(R1,    IN(59), IN(60), R2, 8192, 64, 64,  1);
    gemm(R2,    IN(61), IN(62), RL, 8192, 4,  64,  2);
    alpha_logits_k<<<32, 256, 0, stream>>>(RL, gumbel, LP, out_logits, out_alpha);

    // ---- critic: 115 -> 256 -> 256 -> 1 ----
    gemm(state, IN(63), IN(64), T1, 8192, 256, 115, 1);
    gemm(T1,    IN(65), IN(66), T2, 8192, 256, 256, 1);
    gemm(T2,    IN(67), IN(68), out_value, 8192, 1, 256, 0);
#undef IN
}
